// NaiveCorrelationBlock_45414984188031
// MI455X (gfx1250) — matrix operand order repaired and checked by reading, not compiled, not run on hardware
//
#include <hip/hip_runtime.h>

typedef __attribute__((ext_vector_type(16))) _Float16 v16h;
typedef __attribute__((ext_vector_type(8)))  _Float16 v8h;
typedef __attribute__((ext_vector_type(8)))  float    v8f;

#define C_    128
#define H_    96
#define W_    192
#define HW_   (H_ * W_)         // 18432
#define NB    8
#define DPAD  4
#define ND    81                // (2*D+1)^2
#define MT    32                // w-tile (M) per workgroup
#define HT    4                 // h-tile per workgroup
#define BCOL  48                // staged x2 columns: w0-4 .. w0+43
#define LDA   136               // padded LDS row (halves) for x1  [HT*MT][LDA]
#define LDB   136               // padded LDS row (halves) for x2  [BCOL][LDB]

__global__ __launch_bounds__(512)
void corr_wmma_kernel(const float* __restrict__ x1,
                      const float* __restrict__ x2,
                      float* __restrict__ out) {
  __shared__ _Float16 lA[HT * MT * LDA];   // 34816 B
  __shared__ _Float16 lB[BCOL * LDB];      // 13056 B

  const int wt = blockIdx.x;               // 0..5   (192/32)
  const int h0 = blockIdx.y * HT;          // 0,4,..,92
  const int n  = blockIdx.z;               // 0..7
  const int w0 = wt * MT;

  const int t    = threadIdx.x;            // 0..511 (16 waves)
  const int wave = t >> 5;
  const int lane = t & 31;
  const int hl   = lane >> 4;              // half-lane group
  const int l16  = lane & 15;

  // wave -> (h_local, band pair). pairs: (mt,jt) = (0,0)(0,1)(1,1)(1,2)
  const int h_local = wave >> 2;           // 0..3
  const int pair    = wave & 3;
  const int mt = (pair >= 2) ? 1 : 0;
  const int jt = (pair == 0) ? 0 : ((pair == 3) ? 2 : 1);
  const int hr = h0 + h_local;             // output row of this wave

  // ------- stage x1: HT rows x 32 w x 128 c as f16, layout [(row,m)][c] -----
  {
    const float* x1base = x1 + (size_t)n * (C_ * HW_) + (size_t)h0 * W_ + w0;
    #pragma unroll
    for (int i = 0; i < 8; ++i) {
      int idx = i * 512 + t;               // 0..4095 quads
      int qm  = idx & 7;                   // quad within m
      int row = (idx >> 3) & 3;            // h row
      int c   = idx >> 5;                  // 0..127
      const float4 v = *(const float4*)(x1base + (size_t)c * HW_ +
                                        (size_t)row * W_ + 4 * qm);
      _Float16* dst = lA + (row * MT + 4 * qm) * LDA + c;
      dst[0 * LDA] = (_Float16)v.x;
      dst[1 * LDA] = (_Float16)v.y;
      dst[2 * LDA] = (_Float16)v.z;
      dst[3 * LDA] = (_Float16)v.w;
    }
  }
  __syncthreads();

  // ------- A fragments (16x32 f16 per ISA 7.12.2), reused for all dh --------
  v16h a[4];
  {
    const _Float16* ap = lA + (h_local * MT + 16 * mt + l16) * LDA;
    #pragma unroll
    for (int kk = 0; kk < 4; ++kk) {
      int k0 = kk * 32 + 8 * hl;
      v8h lo = *(const v8h*)(ap + k0);
      v8h hi = *(const v8h*)(ap + k0 + 16);
      #pragma unroll
      for (int e = 0; e < 8; ++e) { a[kk][e] = lo[e]; a[kk][e + 8] = hi[e]; }
    }
  }

  // store geometry: D element (r, lane) -> M = r + 8*hl, N = l16
  const int m_lo  = 16 * mt + 8 * hl;      // + r gives m_abs
  const int n_abs = 16 * jt + l16;         // staged column index
  const int diff  = n_abs - m_lo;          // dw4 at r = 0
  const float* x2n = x2 + (size_t)n * (C_ * HW_);

  for (int it = 0; it < 12; ++it) {        // staged x2 row: hh = h0-4+it
    const int hh = h0 - DPAD + it;
    __syncthreads();                       // previous consumers done

    // ----- stage x2 row hh: 48 cols x 128 c, f16, zero-padded ---------------
    // quads of 4 cols are all-in or all-out of bounds (4 | DPAD, 4 | W)
    {
      const int hc = (hh < 0) ? 0 : ((hh > H_ - 1) ? H_ - 1 : hh);
      const bool hok = (hh >= 0) && (hh < H_);
      const float* rowbase = x2n + (size_t)hc * W_;
      #pragma unroll
      for (int i = 0; i < 3; ++i) {
        int idx = i * 512 + t;             // 0..1535
        int c   = idx / 12;                // 0..127
        int q   = idx - c * 12;            // quad of cols
        int s   = w0 - DPAD + 4 * q;       // quad start wcol (mult of 4)
        int sc  = (s < 0) ? 0 : ((s > W_ - 4) ? W_ - 4 : s);
        float4 v = *(const float4*)(rowbase + (size_t)c * HW_ + sc); // uncond.
        const bool ok = hok && (s >= 0) && (s <= W_ - 4);
        _Float16* dst = lB + (4 * q) * LDB + c;
        dst[0 * LDB] = (_Float16)(ok ? v.x : 0.0f);
        dst[1 * LDB] = (_Float16)(ok ? v.y : 0.0f);
        dst[2 * LDB] = (_Float16)(ok ? v.z : 0.0f);
        dst[3 * LDB] = (_Float16)(ok ? v.w : 0.0f);
      }
    }
    __syncthreads();

    // ----- this wave computes only if dh = hh - hr is in [-4, 4] ------------
    // wave-uniform condition -> EXEC stays all-ones around WMMA
    if (it >= h_local && it <= h_local + 8) {
      const int dhi = it - h_local;        // dh + 4

      v8f acc = {};
      {
        const _Float16* bp = lB + n_abs * LDB;
        #pragma unroll
        for (int kk = 0; kk < 4; ++kk) {
          int kb = kk * 32 + 8 * hl;
          v8h b0 = *(const v8h*)(bp + kb);
          v8h b1 = *(const v8h*)(bp + kb + 16);
          v16h b;
          #pragma unroll
          for (int e = 0; e < 8; ++e) { b[e] = b0[e]; b[e + 8] = b1[e]; }
          acc = __builtin_amdgcn_wmma_f32_16x16x32_f16(
                    false, a[kk], false, b, (short)0, acc, false, false);
        }
      }

      // ----- band diagonals: address moves by (1 - H*W) per r ---------------
      long base_off = (((long)(n * ND + dhi * 9 + diff) * H_) + hr) * W_
                      + w0 + m_lo;
      #pragma unroll
      for (int r = 0; r < 8; ++r) {
        if ((unsigned)(diff - r) <= 8u) {
          out[base_off + (long)r * (1 - HW_)] = acc[r] * (1.0f / 128.0f);
        }
      }
    }
  }
}

extern "C" void kernel_launch(void* const* d_in, const int* in_sizes, int n_in,
                              void* d_out, int out_size, void* d_ws, size_t ws_size,
                              hipStream_t stream) {
  (void)in_sizes; (void)n_in; (void)out_size; (void)d_ws; (void)ws_size;
  const float* x1 = (const float*)d_in[0];
  const float* x2 = (const float*)d_in[1];
  float* out      = (float*)d_out;

  dim3 grid(W_ / MT, H_ / HT, NB);   // (6, 24, 8) = 1152 workgroups
  corr_wmma_kernel<<<grid, 512, 0, stream>>>(x1, x2, out);
}
